// LiquidStateMachine_81003083202809
// MI455X (gfx1250) — matrix operand order repaired and checked by reading, not compiled, not run on hardware
//
#include <hip/hip_runtime.h>

// ---------------------------------------------------------------------------
// Liquid State Machine on MI455X (gfx1250, wave32, WMMA f16 -> f32)
//
// B=2048, D=256, N=512, n_steps=64.
//   ic  = rates @ input_weight^T               (once, WMMA, K=256, streamed)
//   rec = (c>0).f16 @ W.f16                    (per step, WMMA, K=512,
//                                               B-fragments REGISTER-RESIDENT)
//   c   = as*c + (1-as)*(ic + rec)
//   v   = am*v + (1-am)*c
//   ss += (v>0)
// Outputs (concatenated): readout=ss/n, v, c   (each 2048*512 f32)
//
// 128 blocks x 512 threads (16 waves). Each block owns 16 batch rows; each
// wave owns a 16x32 state slice in WMMA C/D fragment layout. Each wave's
// 32-column slice of W (32 x v16h = 256 VGPRs) is loaded ONCE; the 64-step
// loop is pure LDS + v_wmma + VALU. Spikes ping-pong between two LDS
// buffers so only ONE workgroup barrier is needed per step.
// ---------------------------------------------------------------------------

typedef __attribute__((ext_vector_type(16))) _Float16 v16h;
typedef __attribute__((ext_vector_type(8)))  _Float16 v8h;
typedef __attribute__((ext_vector_type(8)))  float    v8f;

#define D_IN   256
#define N_RES  512
#define BN     (2048 * 512)
#define PITCH  520                 // halfs per LDS row (512 + 8 pad -> bank spread)

#define ALPHA_SYN  0.81873075307798182f
#define OM_SYN     0.18126924692201818f
#define ALPHA_MEM  0.95122942450071400f
#define OM_MEM     0.04877057549928600f

// W packed:   32 n-tiles x 16 k-tiles, each fragment = 32 lanes x 16 halfs
// Win packed: 32 n-tiles x  8 k-tiles, same fragment shape
#define WPACK_HALFS   (512 * 512)  // 512 frags * 512 halfs
#define WINPACK_HALFS (256 * 512)  // 256 frags * 512 halfs

__device__ __forceinline__ v16h cat8(v8h a, v8h b) {
    return __builtin_shufflevector(a, b, 0,1,2,3,4,5,6,7,8,9,10,11,12,13,14,15);
}

// ---------------------------------------------------------------------------
// Pack f32 weights into the exact 16-bit WMMA B-fragment lane layout:
//   lane l, half h of fragment (nt,kt):  k = kt*32 + 8*(l>>4) + (h<8 ? h : h+8),
//                                        n = nt*16 + (l&15)
// ---------------------------------------------------------------------------
__global__ __launch_bounds__(256)
void lsm_pack_kernel(const float* __restrict__ W,      // (512,512) row-major [k][n]
                     const float* __restrict__ Win,    // (512,256) row-major [n][d]
                     _Float16* __restrict__ wpack,
                     _Float16* __restrict__ winpack) {
    int id = blockIdx.x * 256 + threadIdx.x;
    if (id < 512 * 32) {                       // reservoir W: 512 fragments
        int lane = id & 31;
        int frag = id >> 5;
        int kt = frag & 15;
        int nt = frag >> 4;
        int n  = nt * 16 + (lane & 15);
        int kb = kt * 32 + ((lane >> 4) << 3);
        v16h vv;
#pragma unroll
        for (int h = 0; h < 16; ++h)
            vv[h] = (_Float16)W[(size_t)(kb + (h < 8 ? h : h + 8)) * N_RES + n];
        *(v16h*)(wpack + (size_t)frag * 512 + lane * 16) = vv;
    } else {                                   // input weight^T: 256 fragments
        id -= 512 * 32;
        if (id >= 256 * 32) return;
        int lane = id & 31;
        int frag = id >> 5;
        int kt = frag & 7;
        int nt = frag >> 3;
        int n  = nt * 16 + (lane & 15);
        int kb = kt * 32 + ((lane >> 4) << 3);
        v16h vv;
#pragma unroll
        for (int h = 0; h < 16; ++h)           // B[k][n] = Win^T[k][n] = Win[n][k]
            vv[h] = (_Float16)Win[(size_t)n * D_IN + kb + (h < 8 ? h : h + 8)];
        *(v16h*)(winpack + (size_t)frag * 512 + lane * 16) = vv;
    }
}

// ---------------------------------------------------------------------------
// Main LSM kernel: 128 blocks x 512 threads (16 waves), W register-resident,
// ping-pong spike buffers (one barrier per step).
// ---------------------------------------------------------------------------
__global__ __launch_bounds__(512)
void lsm_step_kernel(const float* __restrict__ rates,   // (2048,256)
                     const float* __restrict__ vmem0,   // (2048,512)
                     const float* __restrict__ cur0,    // (2048,512)
                     const _Float16* __restrict__ wpack,
                     const _Float16* __restrict__ winpack,
                     const int* __restrict__ nsteps_p,
                     float* __restrict__ out) {
    __shared__ _Float16 lds[2][16 * PITCH];   // ping-pong spike buffers

    const int tid   = threadIdx.x;
    const int lane  = tid & 31;
    const int wave  = tid >> 5;                // 0..15, owns columns [32w, 32w+32)
    const int nLane = lane & 15;               // column within a 16-wide N tile
    const int hiHlf = (lane >> 4) << 3;        // 8*(lane/16): row offset / A k-chunk
    const int rowTile = blockIdx.x;            // 16 batch rows per block
    const int nsteps  = *nsteps_p;

    // ---- stage rates tile (16 x 256) into LDS buf0 as f16 (pitch 520)
#pragma unroll
    for (int i = 0; i < 8; ++i) {
        int e = tid + i * 512;                 // 4096 elements
        int m = e >> 8, d = e & 255;
        lds[0][m * PITCH + d] = (_Float16)rates[(size_t)(rowTile * 16 + m) * D_IN + d];
    }
    __syncthreads();

    // ---- ic = rates @ Win^T for this wave's 16x32 slice (K = 256, 8 k-tiles)
    v8f ic[2] = {};
#pragma unroll
    for (int kt = 0; kt < 8; ++kt) {
        const _Float16* ab = &lds[0][nLane * PITCH + kt * 32 + hiHlf];
        v16h a = cat8(*(const v8h*)ab, *(const v8h*)(ab + 16));
#pragma unroll
        for (int t = 0; t < 2; ++t) {
            int nt = wave * 2 + t;
            v16h b = *(const v16h*)(winpack + ((size_t)(nt * 8 + kt)) * 512 + lane * 16);
            ic[t] = __builtin_amdgcn_wmma_f32_16x16x32_f16(
                        false, a, false, b, (short)0, ic[t], false, false);
        }
    }
    __syncthreads();   // all IC reads of buf0 done before spike writes

    // ---- preload this wave's W slice into registers: 2 nt x 16 kt fragments
    //      (32 x v16h = 256 VGPRs; invariant across all 64 steps)
    v16h bfr[2][16];
#pragma unroll
    for (int t = 0; t < 2; ++t)
#pragma unroll
        for (int kt = 0; kt < 16; ++kt) {
            int nt = wave * 2 + t;
            bfr[t][kt] = *(const v16h*)(wpack + ((size_t)(nt * 16 + kt)) * 512
                                        + lane * 16);
        }

    // ---- load initial state (v_mem, current are inputs; zeros in practice)
    float cS[2][8], vS[2][8], ss[2][8];
#pragma unroll
    for (int t = 0; t < 2; ++t)
#pragma unroll
        for (int r = 0; r < 8; ++r) {
            int row = rowTile * 16 + r + hiHlf;
            int n   = wave * 32 + t * 16 + nLane;
            size_t idx = (size_t)row * N_RES + n;
            cS[t][r] = cur0[idx];
            vS[t][r] = vmem0[idx];
            ss[t][r] = 0.0f;
        }

    // ---- prologue: spikes for step 0 into buf0
#pragma unroll
    for (int t = 0; t < 2; ++t)
#pragma unroll
        for (int r = 0; r < 8; ++r) {
            int m = r + hiHlf;
            int n = wave * 32 + t * 16 + nLane;
            lds[0][m * PITCH + n] = (cS[t][r] > 0.0f) ? (_Float16)1.0f
                                                      : (_Float16)0.0f;
        }
    __syncthreads();

    // ---- time loop: read buf[s&1], write next spikes to buf[1-(s&1)].
    //      One barrier per step; zero global traffic.
    for (int s = 0; s < nsteps; ++s) {
        const _Float16* rbuf = lds[s & 1];
        _Float16*       wbuf = lds[(s & 1) ^ 1];

        // rec = spikes @ W : 16 k-tiles x 2 n-tiles, B from registers
        v8f acc[2] = {};
#pragma unroll
        for (int kt = 0; kt < 16; ++kt) {
            const _Float16* ab = &rbuf[nLane * PITCH + kt * 32 + hiHlf];
            v16h a = cat8(*(const v8h*)ab, *(const v8h*)(ab + 16));
#pragma unroll
            for (int t = 0; t < 2; ++t) {
                acc[t] = __builtin_amdgcn_wmma_f32_16x16x32_f16(
                             false, a, false, bfr[t][kt], (short)0, acc[t],
                             false, false);
            }
        }

        // LIF state update + next step's spikes into the other buffer
#pragma unroll
        for (int t = 0; t < 2; ++t)
#pragma unroll
            for (int r = 0; r < 8; ++r) {
                float cc = ALPHA_SYN * cS[t][r] + OM_SYN * (ic[t][r] + acc[t][r]);
                float vv = ALPHA_MEM * vS[t][r] + OM_MEM * cc;
                cS[t][r] = cc;
                vS[t][r] = vv;
                ss[t][r] += (vv > 0.0f) ? 1.0f : 0.0f;

                int m = r + hiHlf;
                int n = wave * 32 + t * 16 + nLane;
                wbuf[m * PITCH + n] = (cc > 0.0f) ? (_Float16)1.0f
                                                  : (_Float16)0.0f;
            }

        __syncthreads();   // next spikes visible; prior reads complete
    }

    // ---- write outputs: readout, v, c (concatenated flat)
    const float inv = 1.0f / (float)nsteps;
#pragma unroll
    for (int t = 0; t < 2; ++t)
#pragma unroll
        for (int r = 0; r < 8; ++r) {
            int row = rowTile * 16 + r + hiHlf;
            int n   = wave * 32 + t * 16 + nLane;
            size_t idx = (size_t)row * N_RES + n;
            out[idx]          = ss[t][r] * inv;
            out[BN + idx]     = vS[t][r];
            out[2 * BN + idx] = cS[t][r];
        }
}

// ---------------------------------------------------------------------------
// Launch: pack weights into d_ws (768 KB), then run the fused LSM kernel.
// ---------------------------------------------------------------------------
extern "C" void kernel_launch(void* const* d_in, const int* in_sizes, int n_in,
                              void* d_out, int out_size, void* d_ws, size_t ws_size,
                              hipStream_t stream) {
    const float* rates        = (const float*)d_in[0];  // (2048,256)
    const float* input_weight = (const float*)d_in[1];  // (512,256)
    const float* W            = (const float*)d_in[2];  // (512,512)
    const float* vmem0        = (const float*)d_in[3];  // (2048,512)
    const float* cur0         = (const float*)d_in[4];  // (2048,512)
    const int*   nsteps       = (const int*)d_in[5];    // scalar 64

    _Float16* wpack   = (_Float16*)d_ws;
    _Float16* winpack = wpack + WPACK_HALFS;

    lsm_pack_kernel<<<96, 256, 0, stream>>>(W, input_weight, wpack, winpack);

    lsm_step_kernel<<<128, 512, 0, stream>>>(rates, vmem0, cur0,
                                             wpack, winpack, nsteps,
                                             (float*)d_out);
}
